// LEHD_6330781794647
// MI455X (gfx1250) — compile-verified, matrix operand order checked by reading, not run on hardware
//
#include <hip/hip_runtime.h>
#include <hip/hip_bf16.h>

// ---------------------------------------------------------------------------
// LEHD transformer forward for MI455X (gfx1250), wave32 + WMMA f16->f32.
// All matmuls lower to v_wmma_f32_16x16x32_f16.
// ---------------------------------------------------------------------------

typedef __attribute__((ext_vector_type(16))) _Float16 v16h;
typedef __attribute__((ext_vector_type(8)))  _Float16 v8h;
typedef __attribute__((ext_vector_type(8)))  float    v8f;

#define NB    32
#define NN    512
#define ND    256
#define NH    8
#define NL    6
#define MROWS (NB*NN)          /* 16384 */

union FragU { v16h v; v8h h[2]; };

// Load a 16x32 f16 fragment (A or B operand) in the CDNA5 WMMA VGPR striping:
// lane l: row = base + (l&15); halves[0..7]  = K = kk + (l>>4)*8 + 0..7
//                              halves[8..15] = K = kk + 16 + (l>>4)*8 + 0..7
__device__ __forceinline__ v16h load_frag(const _Float16* __restrict__ base,
                                          int row, int ld, int kk, int hs) {
  FragU f;
  const _Float16* p = base + (size_t)row * ld + kk + hs * 8;
  f.h[0] = *(const v8h*)p;
  f.h[1] = *(const v8h*)(p + 16);
  return f.v;
}

#define WMMA(a,b,c) __builtin_amdgcn_wmma_f32_16x16x32_f16(false,(a),false,(b),(short)0,(c),false,false)

// ---------------------------------------------------------------------------
// Generic GEMM: C(M x N) = A(M x 256) @ W(N x 256, ld=ldw)^T + bias [+resid][relu]
// Block: 256 threads = 8 waves (4 M x 2 N), wave tile 32x32, block tile 128x64.
// ---------------------------------------------------------------------------
__global__ __launch_bounds__(256) void gemm_f16(
    const _Float16* __restrict__ A, const _Float16* __restrict__ W, int ldw,
    const float* __restrict__ bias, const float* __restrict__ resid,
    float* __restrict__ outF, _Float16* __restrict__ outH,
    int N, int relu)
{
  const int K = 256;
  int lane = threadIdx.x & 31, wv = threadIdx.x >> 5;
  int nblk = N >> 6;
  int bx = blockIdx.x % nblk, by = blockIdx.x / nblk;
  int m0 = by * 128 + (wv >> 1) * 32;
  int n0 = bx * 64 + (wv & 1) * 32;
  int hs = lane >> 4, li = lane & 15;

  v8f c00 = {}, c01 = {}, c10 = {}, c11 = {};
  for (int kk = 0; kk < K; kk += 32) {
    v16h a0 = load_frag(A, m0 + li,      K,   kk, hs);
    v16h a1 = load_frag(A, m0 + 16 + li, K,   kk, hs);
    v16h b0 = load_frag(W, n0 + li,      ldw, kk, hs);
    v16h b1 = load_frag(W, n0 + 16 + li, ldw, kk, hs);
    c00 = WMMA(a0, b0, c00);
    c01 = WMMA(a0, b1, c01);
    c10 = WMMA(a1, b0, c10);
    c11 = WMMA(a1, b1, c11);
  }
  #pragma unroll
  for (int t = 0; t < 4; ++t) {
    v8f c = (t == 0) ? c00 : (t == 1) ? c01 : (t == 2) ? c10 : c11;
    int mt = m0 + (t >> 1) * 16;
    int col = n0 + (t & 1) * 16 + li;
    float bv = bias ? bias[col] : 0.0f;
    #pragma unroll
    for (int r = 0; r < 8; ++r) {
      int row = mt + r + hs * 8;
      float v = c[r] + bv;
      if (resid) v += resid[(size_t)row * N + col];
      if (relu)  v = fmaxf(v, 0.0f);
      if (outF)  outF[(size_t)row * N + col] = v;
      if (outH)  outH[(size_t)row * N + col] = (_Float16)v;
    }
  }
}

// ---------------------------------------------------------------------------
// LayerNorm: one wave per 256-wide row, f32 in -> f16 out.
// ---------------------------------------------------------------------------
__global__ __launch_bounds__(256) void ln_f16out(
    const float* __restrict__ in, const float* __restrict__ g,
    const float* __restrict__ be, _Float16* __restrict__ out)
{
  int lane = threadIdx.x & 31, wv = threadIdx.x >> 5;
  size_t row = (size_t)blockIdx.x * 8 + wv;
  const float* x = in + row * ND;
  const float4* xv = (const float4*)(x + lane * 8);
  float4 f0 = xv[0], f1 = xv[1];
  float e[8] = { f0.x, f0.y, f0.z, f0.w, f1.x, f1.y, f1.z, f1.w };
  float s = 0.f;
  #pragma unroll
  for (int i = 0; i < 8; ++i) s += e[i];
  #pragma unroll
  for (int o = 16; o >= 1; o >>= 1) s += __shfl_xor(s, o);
  float mean = s * (1.0f / 256.0f);
  float vs = 0.f;
  #pragma unroll
  for (int i = 0; i < 8; ++i) { float d = e[i] - mean; vs += d * d; }
  #pragma unroll
  for (int o = 16; o >= 1; o >>= 1) vs += __shfl_xor(vs, o);
  float inv = rsqrtf(vs * (1.0f / 256.0f) + 1e-5f);
  int c0 = lane * 8;
  v8h o8;
  #pragma unroll
  for (int i = 0; i < 8; ++i)
    o8[i] = (_Float16)((e[i] - mean) * inv * g[c0 + i] + be[c0 + i]);
  *(v8h*)(out + row * ND + c0) = o8;
}

// ---------------------------------------------------------------------------
// Embedding gather (f32 table -> f16 rows).
// ---------------------------------------------------------------------------
__global__ void gather_embed(const float* __restrict__ sol,
                             const float* __restrict__ embds,
                             _Float16* __restrict__ out)
{
  int row = blockIdx.x, c = threadIdx.x;
  int idx = (int)sol[(size_t)row * 4];
  out[(size_t)row * ND + c] = (_Float16)embds[(size_t)idx * ND + c];
}

// nd feature: 2*((sol[:,:,2]-sol[:,0,3])/cap), nd[:,0]=0
__global__ void nd_kernel(const float* __restrict__ sol,
                          const float* __restrict__ cap,
                          float* __restrict__ nd)
{
  int r = blockIdx.x * 256 + threadIdx.x;
  int b = r >> 9, n = r & 511;
  float rc = sol[((size_t)b * NN) * 4 + 3];
  float v = 2.0f * ((sol[(size_t)r * 4 + 2] - rc) / cap[b]);
  nd[r] = (n == 0) ? 0.0f : v;
}

// ---------------------------------------------------------------------------
// Weight preparation: effective f16 weights.
// ---------------------------------------------------------------------------
__global__ void prep_axpy(const float* __restrict__ W, const float* __restrict__ R,
                          _Float16* __restrict__ out, int n)
{
  for (int i = blockIdx.x * 256 + threadIdx.x; i < n; i += gridDim.x * 256)
    out[i] = (_Float16)(W[i] + 0.1f * R[i]);
}
__global__ void prep_cvt(const float* __restrict__ W, _Float16* __restrict__ out, int n)
{
  for (int i = blockIdx.x * 256 + threadIdx.x; i < n; i += gridDim.x * 256)
    out[i] = (_Float16)W[i];
}
// ne 'cand' effective weights: (1-a)Wc + a(Wsw + 0.1Rsw), a = sigmoid(iw).
// Splits K=256 part (wcK, ld=256) from column 256 (wcLast) + effective bias.
__global__ void prep_ne(const float* __restrict__ Wc, const float* __restrict__ bcp,
                        const float* __restrict__ Wsw, const float* __restrict__ bsw,
                        const float* __restrict__ Rsw, const float* __restrict__ iw,
                        _Float16* __restrict__ wcK, float* __restrict__ wcLast,
                        float* __restrict__ bcEff)
{
  int i = blockIdx.x * 256 + threadIdx.x;   // grid 257 blocks -> exactly 65792
  float a = 1.0f / (1.0f + __expf(-iw[0]));
  int n = i / 257, k = i % 257;
  float eff = (1.0f - a) * Wc[i] + a * (Wsw[i] + 0.1f * Rsw[i]);
  if (k < 256) wcK[n * 256 + k] = (_Float16)eff;
  else         wcLast[n] = eff;
  if (i < 256) bcEff[i] = (1.0f - a) * bcp[i] + a * bsw[i];
}

// ---------------------------------------------------------------------------
// ne epilogue: rank-1 nd update for normal rows; full recompute for the
// 3 special positions per batch (src n==0, dst n==510, dep n==511).
// ---------------------------------------------------------------------------
__global__ void ne_epilogue(const _Float16* __restrict__ x, const float* __restrict__ nd,
                            float* __restrict__ h,
                            const float* __restrict__ Ws,  const float* __restrict__ bs,
                            const float* __restrict__ Wd,  const float* __restrict__ bd,
                            const float* __restrict__ Wdep,const float* __restrict__ bdep,
                            const float* __restrict__ Wsw, const float* __restrict__ bsw,
                            const float* __restrict__ Rsw, const float* __restrict__ iw,
                            const float* __restrict__ wcLast)
{
  int r = blockIdx.x, n = threadIdx.x;
  int pos = r & 511;
  float ndv = nd[r];
  if (pos != 0 && pos != 510 && pos != 511) {
    h[(size_t)r * ND + n] += ndv * wcLast[n];
  } else {
    const float *W, *bseg;
    if (pos == 0)        { W = Ws;   bseg = bs;   }
    else if (pos == 510) { W = Wd;   bseg = bd;   }
    else                 { W = Wdep; bseg = bdep; }
    float a = 1.0f / (1.0f + __expf(-iw[0]));
    const _Float16* xr = x + (size_t)r * ND;
    float acc = 0.0f;
    for (int k = 0; k < 256; ++k) {
      float we = (1.0f - a) * W[n * 257 + k] + a * (Wsw[n * 257 + k] + 0.1f * Rsw[n * 257 + k]);
      acc += (float)xr[k] * we;
    }
    float wl = (1.0f - a) * W[n * 257 + 256] + a * (Wsw[n * 257 + 256] + 0.1f * Rsw[n * 257 + 256]);
    acc += ndv * wl;
    acc += (1.0f - a) * bseg[n] + a * bsw[n];
    h[(size_t)r * ND + n] = acc;
  }
}

// ---------------------------------------------------------------------------
// QKV reshape: (B*N,768) f16 -> Qh/Kh (B,H,N,32) and V^T (B,H,32,N).
// Q pre-scaled by 1/sqrt(dh).
// ---------------------------------------------------------------------------
__global__ void qkv_split(const _Float16* __restrict__ qkv,
                          _Float16* __restrict__ Qh, _Float16* __restrict__ Kh,
                          _Float16* __restrict__ VTh)
{
  int r = blockIdx.x, c = threadIdx.x;            // r: 0..16383, c: 0..255
  int b = r >> 9, n = r & 511;
  int hd = c >> 5, dh = c & 31;
  size_t head = (size_t)(b * NH + hd);
  const _Float16* src = qkv + (size_t)r * 768;
  float q = (float)src[c] * 0.17677669529663687f;  // 1/sqrt(32)
  Qh[(head * NN + n) * 32 + dh] = (_Float16)q;
  Kh[(head * NN + n) * 32 + dh] = src[256 + c];
  VTh[head * 32 * NN + (size_t)dh * NN + n] = src[512 + c];
}

// ---------------------------------------------------------------------------
// Attention: one wave per (batch, head, 16-query tile). Online softmax.
// S = Q(16x32) @ K^T via WMMA; P repacked C->A layout via LDS; O += P @ V^T.
// ---------------------------------------------------------------------------
__global__ __launch_bounds__(256) void attn_kernel(
    const _Float16* __restrict__ Qh, const _Float16* __restrict__ Kh,
    const _Float16* __restrict__ VTh, _Float16* __restrict__ attn)
{
  __shared__ _Float16 pbuf[8][16 * 32];
  int lane = threadIdx.x & 31, wv = threadIdx.x >> 5;
  int wid = blockIdx.x * 8 + wv;                  // 0..8191
  int qt = wid & 31;                              // query tile
  int bh = wid >> 5;                              // 0..255 (b*8+h)
  int hs = lane >> 4, li = lane & 15;
  const _Float16* Qb = Qh + (size_t)bh * NN * 32;
  const _Float16* Kb = Kh + (size_t)bh * NN * 32;
  const _Float16* Vb = VTh + (size_t)bh * 32 * NN;

  v16h qf = load_frag(Qb, qt * 16 + li, 32, 0, hs);
  float mrow[8], srow[8];
  #pragma unroll
  for (int r = 0; r < 8; ++r) { mrow[r] = -1e30f; srow[r] = 0.0f; }
  v8f o0 = {}, o1 = {};

  for (int kb = 0; kb < NN; kb += 32) {
    v16h bk0 = load_frag(Kb, kb + li,      32, 0, hs);
    v16h bk1 = load_frag(Kb, kb + 16 + li, 32, 0, hs);
    v8f z0 = {}, z1 = {};
    v8f s0 = WMMA(qf, bk0, z0);
    v8f s1 = WMMA(qf, bk1, z1);
    #pragma unroll
    for (int r = 0; r < 8; ++r) {
      float tm = fmaxf(s0[r], s1[r]);
      #pragma unroll
      for (int o = 8; o >= 1; o >>= 1) tm = fmaxf(tm, __shfl_xor(tm, o));
      float mn = fmaxf(mrow[r], tm);
      float sc = __expf(mrow[r] - mn);
      float p0 = __expf(s0[r] - mn);
      float p1 = __expf(s1[r] - mn);
      float rs = p0 + p1;
      #pragma unroll
      for (int o = 8; o >= 1; o >>= 1) rs += __shfl_xor(rs, o);
      srow[r] = srow[r] * sc + rs;
      mrow[r] = mn;
      o0[r] *= sc; o1[r] *= sc;
      int mr = r + hs * 8;
      pbuf[wv][mr * 32 + li]      = (_Float16)p0;
      pbuf[wv][mr * 32 + 16 + li] = (_Float16)p1;
    }
    asm volatile("s_wait_dscnt 0" ::: "memory");
    FragU pf;
    const _Float16* pp = &pbuf[wv][li * 32 + hs * 8];
    pf.h[0] = *(const v8h*)pp;
    pf.h[1] = *(const v8h*)(pp + 16);
    v16h bv0 = load_frag(Vb, li,      NN, kb, hs);
    v16h bv1 = load_frag(Vb, 16 + li, NN, kb, hs);
    o0 = WMMA(pf.v, bv0, o0);
    o1 = WMMA(pf.v, bv1, o1);
  }
  int b = bh >> 3, hd = bh & 7;
  #pragma unroll
  for (int r = 0; r < 8; ++r) {
    float inv = 1.0f / srow[r];
    size_t row = (size_t)(b << 9) + qt * 16 + r + hs * 8;
    attn[row * ND + hd * 32 + li]      = (_Float16)(o0[r] * inv);
    attn[row * ND + hd * 32 + 16 + li] = (_Float16)(o1[r] * inv);
  }
}

// ---------------------------------------------------------------------------
// Pointer head + output concat: out[b,j]=logit0(n=1+j), out[b,509+j]=logit1.
// One wave per (b, j).
// ---------------------------------------------------------------------------
__global__ __launch_bounds__(256) void ptr_kernel(
    const _Float16* __restrict__ hdec, const float* __restrict__ Wp,
    const float* __restrict__ bp, float* __restrict__ out)
{
  int wid = blockIdx.x * 8 + (threadIdx.x >> 5);
  if (wid >= NB * 509) return;
  int lane = threadIdx.x & 31;
  int b = wid / 509, j = wid % 509;
  const _Float16* hr = hdec + ((size_t)b * NN + 1 + j) * ND;
  float d0 = 0.f, d1 = 0.f;
  #pragma unroll
  for (int i = 0; i < 8; ++i) {
    int k = lane * 8 + i;
    float hv = (float)hr[k];
    d0 += hv * Wp[k];
    d1 += hv * Wp[256 + k];
  }
  #pragma unroll
  for (int o = 16; o >= 1; o >>= 1) { d0 += __shfl_xor(d0, o); d1 += __shfl_xor(d1, o); }
  if (lane == 0) {
    out[(size_t)b * 1018 + j]       = d0 + bp[0];
    out[(size_t)b * 1018 + 509 + j] = d1 + bp[1];
  }
}

// ---------------------------------------------------------------------------
// Host orchestration.
// Input flattening assumed: depth-first dict-insertion order of setup_inputs().
// ---------------------------------------------------------------------------
static void launch_gemm(const _Float16* A, const _Float16* W, int ldw,
                        const float* bias, const float* resid,
                        float* oF, _Float16* oH, int N, int relu, hipStream_t s)
{
  dim3 grid((N / 64) * (MROWS / 128));
  gemm_f16<<<grid, 256, 0, s>>>(A, W, ldw, bias, resid, oF, oH, N, relu);
}

extern "C" void kernel_launch(void* const* d_in, const int* in_sizes, int n_in,
                              void* d_out, int out_size, void* d_ws, size_t ws_size,
                              hipStream_t stream)
{
  (void)in_sizes; (void)n_in; (void)out_size; (void)ws_size;
  const float* sol   = (const float*)d_in[0];
  const float* cap   = (const float*)d_in[1];
  const float* embds = (const float*)d_in[2];
  const float* Wt    = (const float*)d_in[3];
  const float* bt    = (const float*)d_in[4];
  const float* Rt    = (const float*)d_in[5];
  const float* eg    = (const float*)d_in[6];
  const float* ebe   = (const float*)d_in[7];
  const float* Wg    = (const float*)d_in[8];
  const float* bg    = (const float*)d_in[9];
  const float* Ws    = (const float*)d_in[10];
  const float* bs    = (const float*)d_in[11];
  const float* Wd    = (const float*)d_in[12];
  const float* bd    = (const float*)d_in[13];
  const float* Wdep  = (const float*)d_in[14];
  const float* bdep  = (const float*)d_in[15];
  const float* Wc    = (const float*)d_in[16];
  const float* bc    = (const float*)d_in[17];
  const float* Wsw   = (const float*)d_in[18];
  const float* bsw   = (const float*)d_in[19];
  const float* Rsw   = (const float*)d_in[20];
  const float* iw    = (const float*)d_in[21];
  const float* dgg   = (const float*)d_in[88];
  const float* dgbe  = (const float*)d_in[89];
  const float* Wdec  = (const float*)d_in[90];
  const float* bdec  = (const float*)d_in[91];
  const float* Rdec  = (const float*)d_in[92];
  const float* Wp    = (const float*)d_in[93];
  const float* bp    = (const float*)d_in[94];

  // ---- workspace layout -------------------------------------------------
  char* ws = (char*)d_ws;
  size_t off = 0;
  auto take = [&](size_t bytes) -> char* {
    off = (off + 255) & ~(size_t)255;
    char* p = ws + off; off += bytes; return p;
  };
  _Float16* embH = (_Float16*)take((size_t)MROWS * ND * 2);
  float*    hbuf = (float*)   take((size_t)MROWS * ND * 4);
  _Float16* tbuf = (_Float16*)take((size_t)MROWS * ND * 2);
  _Float16* xbuf = (_Float16*)take((size_t)MROWS * ND * 2);
  float*    ndb  = (float*)   take((size_t)MROWS * 4);
  _Float16* qkvb = (_Float16*)take((size_t)MROWS * 3 * ND * 2);
  _Float16* attnb = qkvb;                              // alias (qkv dead after split)
  _Float16* Qh   = (_Float16*)take((size_t)MROWS * ND * 2);
  _Float16* Kh   = (_Float16*)take((size_t)MROWS * ND * 2);
  _Float16* VTh  = (_Float16*)take((size_t)MROWS * ND * 2);
  _Float16* hdec = Qh;                                 // alias (Qh dead after attn)
  _Float16* wtE  = (_Float16*)take(65536 * 2);
  _Float16* wgH  = (_Float16*)take(65536 * 2);
  _Float16* wcK  = (_Float16*)take(65536 * 2);
  float*    wcLast = (float*) take(256 * 4);
  float*    bcEff  = (float*) take(256 * 4);
  _Float16* wqkvH[NL]; _Float16* woH[NL]; _Float16* wfE[NL];
  for (int l = 0; l < NL; ++l) {
    wqkvH[l] = (_Float16*)take(196608 * 2);
    woH[l]   = (_Float16*)take(65536 * 2);
    wfE[l]   = (_Float16*)take(65536 * 2);
  }
  _Float16* wdecH = (_Float16*)take(65536 * 2);

  // ---- weight prep ------------------------------------------------------
  prep_axpy<<<256, 256, 0, stream>>>(Wt, Rt, wtE, 65536);
  prep_cvt <<<256, 256, 0, stream>>>(Wg, wgH, 65536);
  prep_ne  <<<257, 256, 0, stream>>>(Wc, bc, Wsw, bsw, Rsw, iw, wcK, wcLast, bcEff);
  for (int l = 0; l < NL; ++l) {
    const float* Wqkv = (const float*)d_in[22 + l * 11 + 4];
    const float* Wo   = (const float*)d_in[22 + l * 11 + 6];
    const float* Wf   = (const float*)d_in[22 + l * 11 + 8];
    const float* Rf   = (const float*)d_in[22 + l * 11 + 10];
    prep_cvt <<<256, 256, 0, stream>>>(Wqkv, wqkvH[l], 196608);
    prep_cvt <<<256, 256, 0, stream>>>(Wo, woH[l], 65536);
    prep_axpy<<<256, 256, 0, stream>>>(Wf, Rf, wfE[l], 65536);
  }
  prep_axpy<<<256, 256, 0, stream>>>(Wdec, Rdec, wdecH, 65536);

  // ---- encoder front ----------------------------------------------------
  gather_embed<<<MROWS, 256, 0, stream>>>(sol, embds, embH);
  launch_gemm(embH, wtE, 256, bt, nullptr, hbuf, nullptr, 256, 0, stream);   // trans _sw
  ln_f16out<<<MROWS / 8, 256, 0, stream>>>(hbuf, eg, ebe, tbuf);
  launch_gemm(tbuf, wgH, 256, bg, nullptr, nullptr, xbuf, 256, 1, stream);   // enc relu
  nd_kernel<<<MROWS / 256, 256, 0, stream>>>(sol, cap, ndb);
  launch_gemm(xbuf, wcK, 256, bcEff, nullptr, hbuf, nullptr, 256, 0, stream);// ne (cand)
  ne_epilogue<<<MROWS, 256, 0, stream>>>(xbuf, ndb, hbuf, Ws, bs, Wd, bd,
                                         Wdep, bdep, Wsw, bsw, Rsw, iw, wcLast);

  // ---- transformer layers ----------------------------------------------
  for (int l = 0; l < NL; ++l) {
    const float* ln1g = (const float*)d_in[22 + l * 11 + 0];
    const float* ln1b = (const float*)d_in[22 + l * 11 + 1];
    const float* ln2g = (const float*)d_in[22 + l * 11 + 2];
    const float* ln2b = (const float*)d_in[22 + l * 11 + 3];
    const float* bqkv = (const float*)d_in[22 + l * 11 + 5];
    const float* bo   = (const float*)d_in[22 + l * 11 + 7];
    const float* bf   = (const float*)d_in[22 + l * 11 + 9];

    ln_f16out<<<MROWS / 8, 256, 0, stream>>>(hbuf, ln1g, ln1b, tbuf);
    launch_gemm(tbuf, wqkvH[l], 256, bqkv, nullptr, nullptr, qkvb, 768, 0, stream);
    qkv_split<<<MROWS, 256, 0, stream>>>(qkvb, Qh, Kh, VTh);
    attn_kernel<<<(NB * NH * 32) / 8, 256, 0, stream>>>(Qh, Kh, VTh, attnb);
    launch_gemm(attnb, woH[l], 256, bo, hbuf, hbuf, nullptr, 256, 0, stream); // +resid
    ln_f16out<<<MROWS / 8, 256, 0, stream>>>(hbuf, ln2g, ln2b, tbuf);
    launch_gemm(tbuf, wfE[l], 256, bf, hbuf, hbuf, nullptr, 256, 0, stream);  // +resid
  }

  // ---- decoder head -----------------------------------------------------
  ln_f16out<<<MROWS / 8, 256, 0, stream>>>(hbuf, dgg, dgbe, tbuf);
  launch_gemm(tbuf, wdecH, 256, bdec, nullptr, nullptr, hdec, 256, 1, stream);
  ptr_kernel<<<(NB * 509 + 7) / 8, 256, 0, stream>>>(hdec, Wp, bp, (float*)d_out);
}
